// StTokenizer_37847251812634
// MI455X (gfx1250) — matrix operand order repaired and checked by reading, not compiled, not run on hardware
//
#include <hip/hip_runtime.h>
#include <string.h>

// ---------------------------------------------------------------------------
// Problem constants (from reference)
// ---------------------------------------------------------------------------
#define CB   8        // batch
#define CL   128      // sequence length
#define CN   20000    // nodes
#define CM   60000    // edges
#define CT   288      // time steps
#define CS   6        // time features
#define CF   8        // static feature fields
#define CV   32       // vocab per field
#define CD   128      // embedding dim
#define CH   2        // GAT heads
#define CDM  768      // output dim

typedef __bf16 bf16_t;
typedef __bf16 v16bf __attribute__((ext_vector_type(16)));
typedef __bf16 v8bf  __attribute__((ext_vector_type(8)));
typedef float  v8f   __attribute__((ext_vector_type(8)));

// convert two float4 into 8 consecutive bf16 elements of a v16bf
__device__ __forceinline__ void cvt8(v16bf& v, int base, const float4& x, const float4& y)
{
    v[base + 0] = (bf16_t)x.x; v[base + 1] = (bf16_t)x.y;
    v[base + 2] = (bf16_t)x.z; v[base + 3] = (bf16_t)x.w;
    v[base + 4] = (bf16_t)y.x; v[base + 5] = (bf16_t)y.y;
    v[base + 6] = (bf16_t)y.z; v[base + 7] = (bf16_t)y.w;
}

// ---------------------------------------------------------------------------
// GEMM, B pre-packed to bf16 in WMMA B-operand layout Bp[n][Kpad] (K
// contiguous per column). A fp32 row-major. C = act(A*B + bias).
// One wave computes a 16x64 tile: one A fragment is reused by 4 WMMAs per
// K-step (4x less A traffic). Requires: M %16==0, N %64==0, K %32==0,
// lda %4==0. 8 waves/block, batched via blockIdx.y.
// ---------------------------------------------------------------------------
__global__ void gemm_bpk(const float* __restrict__ A,
                         const bf16_t* __restrict__ Bp,
                         const float* __restrict__ bias,
                         float* __restrict__ C,
                         int M, int N, int K,
                         int lda, int ldc,
                         long long sA, long long sB, long long sC,
                         int act)
{
    int ntn = N >> 6;                       // 64-wide N tiles
    int totalTiles = (M >> 4) * ntn;
    int tile = blockIdx.x * (blockDim.x >> 5) + (threadIdx.x >> 5);
    if (tile >= totalTiles) return;
    int tm = (tile / ntn) << 4;
    int tn = (tile % ntn) << 6;
    long long batch = blockIdx.y;
    A  += batch * sA;
    Bp += batch * sB;
    C  += batch * sC;

    int lane = threadIdx.x & 31;
    int half = lane >> 4;
    int rc   = lane & 15;

    const float*  arow = A + (long long)(tm + rc) * lda + 16 * half;
    const bf16_t* bc   = Bp + (long long)(tn + rc) * K + 16 * half;
    const long long bstep = (long long)16 * K;   // 16 columns apart

    v8f acc0 = {0.f,0.f,0.f,0.f,0.f,0.f,0.f,0.f};
    v8f acc1 = acc0, acc2 = acc0, acc3 = acc0;

    for (int k0 = 0; k0 < K; k0 += 32) {
        // A 16x32 bf16 layout: lanes0-15 K = 0-15; lanes16-31 K = 16-31
        const float4* ap0 = (const float4*)(arow + k0);
        const float4* ap1 = (const float4*)(arow + k0 + 8);
        float4 a0 = ap0[0], a1 = ap0[1], a2 = ap1[0], a3 = ap1[1];
        v16bf av;
        cvt8(av, 0, a0, a1);
        cvt8(av, 8, a2, a3);
        // B 32x16 layout: lane = column; lanes0-15 K=0-15, lanes16-31 K=16-31
        const v8bf* bp0 = (const v8bf*)(bc + k0);
        const v8bf* bp1 = (const v8bf*)(bc + bstep + k0);
        const v8bf* bp2 = (const v8bf*)(bc + 2 * bstep + k0);
        const v8bf* bp3 = (const v8bf*)(bc + 3 * bstep + k0);
        v8bf b0l = bp0[0], b0h = bp0[1];
        v8bf b1l = bp1[0], b1h = bp1[1];
        v8bf b2l = bp2[0], b2h = bp2[1];
        v8bf b3l = bp3[0], b3h = bp3[1];
        v16bf bv0 = __builtin_shufflevector(b0l, b0h, 0,1,2,3,4,5,6,7,8,9,10,11,12,13,14,15);
        v16bf bv1 = __builtin_shufflevector(b1l, b1h, 0,1,2,3,4,5,6,7,8,9,10,11,12,13,14,15);
        v16bf bv2 = __builtin_shufflevector(b2l, b2h, 0,1,2,3,4,5,6,7,8,9,10,11,12,13,14,15);
        v16bf bv3 = __builtin_shufflevector(b3l, b3h, 0,1,2,3,4,5,6,7,8,9,10,11,12,13,14,15);
        acc0 = __builtin_amdgcn_wmma_f32_16x16x32_bf16(false, av, false, bv0, (short)0, acc0, false, false);
        acc1 = __builtin_amdgcn_wmma_f32_16x16x32_bf16(false, av, false, bv1, (short)0, acc1, false, false);
        acc2 = __builtin_amdgcn_wmma_f32_16x16x32_bf16(false, av, false, bv2, (short)0, acc2, false, false);
        acc3 = __builtin_amdgcn_wmma_f32_16x16x32_bf16(false, av, false, bv3, (short)0, acc3, false, false);
    }

    v8f accs[4] = {acc0, acc1, acc2, acc3};
#pragma unroll
    for (int j = 0; j < 4; ++j) {
        int col = tn + 16 * j + rc;
        float bb = bias ? bias[col] : 0.f;
#pragma unroll
        for (int r = 0; r < 8; ++r) {
            int row = tm + r + 8 * half;
            float v = accs[j][r] + bb;
            if (act) v = fmaxf(v, 0.f);
            C[(long long)row * ldc + col] = v;
        }
    }
}

// ---------------------------------------------------------------------------
// GEMM with transposed fp32 B (B stored N x K row-major, i.e. C = A * B^T).
// 16x64 tile per wave; B rows contiguous in K. Used for q @ k^T.
// ---------------------------------------------------------------------------
__global__ void gemm_btr(const float* __restrict__ A,
                         const float* __restrict__ Bt,
                         float* __restrict__ C,
                         int M, int N, int K,
                         int lda, int ldb, int ldc,
                         long long sA, long long sB, long long sC)
{
    int ntn = N >> 6;
    int totalTiles = (M >> 4) * ntn;
    int tile = blockIdx.x * (blockDim.x >> 5) + (threadIdx.x >> 5);
    if (tile >= totalTiles) return;
    int tm = (tile / ntn) << 4;
    int tn = (tile % ntn) << 6;
    long long batch = blockIdx.y;
    A  += batch * sA;
    Bt += batch * sB;
    C  += batch * sC;

    int lane = threadIdx.x & 31;
    int half = lane >> 4;
    int rc   = lane & 15;

    const float* arow = A  + (long long)(tm + rc) * lda + 16 * half;
    const float* brow = Bt + (long long)(tn + rc) * ldb + 16 * half;
    const long long bstep = (long long)16 * ldb;

    v8f acc0 = {0.f,0.f,0.f,0.f,0.f,0.f,0.f,0.f};
    v8f acc1 = acc0, acc2 = acc0, acc3 = acc0;

    for (int k0 = 0; k0 < K; k0 += 32) {
        const float4* ap0 = (const float4*)(arow + k0);
        const float4* ap1 = (const float4*)(arow + k0 + 8);
        float4 a0 = ap0[0], a1 = ap0[1], a2 = ap1[0], a3 = ap1[1];
        v16bf av;
        cvt8(av, 0, a0, a1);
        cvt8(av, 8, a2, a3);
#pragma unroll
        for (int j = 0; j < 4; ++j) {
            const float4* bp = (const float4*)(brow + j * bstep + k0);
            float4 b0 = bp[0], b1 = bp[1], b2 = bp[2], b3 = bp[3];
            v16bf bv;
            cvt8(bv, 0, b0, b1);
            cvt8(bv, 8, b2, b3);
            v8f* accp = (j == 0) ? &acc0 : (j == 1) ? &acc1 : (j == 2) ? &acc2 : &acc3;
            *accp = __builtin_amdgcn_wmma_f32_16x16x32_bf16(
                        false, av, false, bv, (short)0, *accp, false, false);
        }
    }

    v8f accs[4] = {acc0, acc1, acc2, acc3};
#pragma unroll
    for (int j = 0; j < 4; ++j) {
        int col = tn + 16 * j + rc;
#pragma unroll
        for (int r = 0; r < 8; ++r) {
            int row = tm + r + 8 * half;
            C[(long long)row * ldc + col] = accs[j][r];
        }
    }
}

// ---------------------------------------------------------------------------
// Pack fp32 row-major B (Kdim x N, ldb = row stride) into bf16 Bp[b][n][Kpad],
// zero-padding K up to Kpad. Batched: per-batch input stride sB elements.
// ---------------------------------------------------------------------------
__global__ void pack_b(const float* __restrict__ B, bf16_t* __restrict__ Bp,
                       int Kdim, int Kpad, int N, int ldb,
                       long long sB, long long tot)
{
    long long i = (long long)blockIdx.x * blockDim.x + threadIdx.x;
    if (i >= tot) return;
    long long per = (long long)N * Kpad;
    int b = (int)(i / per);
    long long r = i % per;
    int n = (int)(r / Kpad);
    int k = (int)(r % Kpad);
    float v = (k < Kdim) ? B[(long long)b * sB + (long long)k * ldb + n] : 0.f;
    Bp[i] = (bf16_t)v;
}

// ---------------------------------------------------------------------------
// Elementwise / gather / fill helpers
// ---------------------------------------------------------------------------
__global__ void fill_u32(unsigned* __restrict__ p, unsigned val, long long n)
{
    long long i = (long long)blockIdx.x * blockDim.x + threadIdx.x;
    if (i < n) p[i] = val;
}

// A_g[n][f*128+d] = emb_tables[f][static_features[n][f]][d]
__global__ void gather_static(const int* __restrict__ sf,
                              const float* __restrict__ emb,
                              float* __restrict__ Ag)
{
    long long i = (long long)blockIdx.x * blockDim.x + threadIdx.x;
    long long tot = (long long)CN * (CF * CD);
    if (i >= tot) return;
    long long node = i >> 10;
    int c = (int)(i & 1023);
    int f = c >> 7, d = c & 127;
    int v = sf[node * CF + f];
    Ag[i] = emb[((long long)f * CV + v) * CD + d];
}

// de0[b*N+n][s] (padded to 32 cols) = dynamic_features[n][btid[b][0]][s], s<6 else 0
__global__ void gather_dyn(const float* __restrict__ dynf,
                           const int* __restrict__ btid,
                           float* __restrict__ de0)
{
    long long i = (long long)blockIdx.x * blockDim.x + threadIdx.x;
    long long tot = (long long)CB * CN * 32;
    if (i >= tot) return;
    long long r = i >> 5;
    int s = (int)(i & 31);
    int b = (int)(r / CN);
    int n = (int)(r % CN);
    float v = 0.f;
    if (s < CS) {
        int tid = btid[b * CL + 0];
        v = dynf[((long long)n * CT + tid) * CS + s];
    }
    de0[i] = v;
}

// dst (rows x colsPad) = src (rows x cols) zero-padded
__global__ void pad_rows(const float* __restrict__ src, float* __restrict__ dst,
                         int cols, int colsPad, long long rows)
{
    long long i = (long long)blockIdx.x * blockDim.x + threadIdx.x;
    long long tot = rows * colsPad;
    if (i >= tot) return;
    long long r = i / colsPad;
    int c = (int)(i % colsPad);
    dst[i] = (c < cols) ? src[r * cols + c] : 0.f;
}

// es/ed per (node, head): dot(h[node, head, :], a_src/a_dst[head, :]); wave reduce
__global__ void gat_scores(const float* __restrict__ feat,
                           const float* __restrict__ asrc,
                           const float* __restrict__ adst,
                           float* __restrict__ es, float* __restrict__ ed, int n)
{
    int node = blockIdx.x * (blockDim.x >> 5) + (threadIdx.x >> 5);
    if (node >= n) return;
    int lane = threadIdx.x & 31;
    for (int hh = 0; hh < CH; ++hh) {
        float s = 0.f, d = 0.f;
        for (int i = lane; i < CD; i += 32) {
            float hv = feat[(long long)node * (CH * CD) + hh * CD + i];
            s += hv * asrc[hh * CD + i];
            d += hv * adst[hh * CD + i];
        }
        for (int o = 16; o > 0; o >>= 1) {
            s += __shfl_xor(s, o, 32);
            d += __shfl_xor(d, o, 32);
        }
        if (lane == 0) {
            es[(long long)node * CH + hh] = s;
            ed[(long long)node * CH + hh] = d;
        }
    }
}

// ordered-int float max transform (involution, preserves float order)
__device__ __forceinline__ int f2ord(float f)
{
    int i = __float_as_int(f);
    return (i >= 0) ? i : (i ^ 0x7fffffff);
}
__device__ __forceinline__ float ord2f(int i)
{
    return __int_as_float((i >= 0) ? i : (i ^ 0x7fffffff));
}

__global__ void edge_e_max(const int* __restrict__ src, const int* __restrict__ dst,
                           const float* __restrict__ es, const float* __restrict__ ed,
                           float* __restrict__ e, int* __restrict__ mx,
                           int Me, int nNodes, long long tot)
{
    long long i = (long long)blockIdx.x * blockDim.x + threadIdx.x;
    if (i >= tot) return;
    int hh = (int)(i & 1);
    long long eid = i >> 1;
    int b  = (int)(eid / Me);
    int ee = (int)(eid % Me);
    int s = src[ee] + b * nNodes;
    int d = dst[ee] + b * nNodes;
    float v = es[(long long)s * CH + hh] + ed[(long long)d * CH + hh];
    v = (v > 0.f) ? v : 0.2f * v;
    e[i] = v;
    atomicMax(&mx[(long long)d * CH + hh], f2ord(v));
}

__global__ void edge_exp(const int* __restrict__ dst,
                         float* __restrict__ e, const int* __restrict__ mx,
                         float* __restrict__ den,
                         int Me, int nNodes, long long tot)
{
    long long i = (long long)blockIdx.x * blockDim.x + threadIdx.x;
    if (i >= tot) return;
    int hh = (int)(i & 1);
    long long eid = i >> 1;
    int b  = (int)(eid / Me);
    int ee = (int)(eid % Me);
    int d = dst[ee] + b * nNodes;
    float m = fmaxf(ord2f(mx[(long long)d * CH + hh]), -1e30f);
    float ex = __expf(e[i] - m);
    e[i] = ex;
    atomicAdd(&den[(long long)d * CH + hh], ex);
}

__global__ void edge_agg(const int* __restrict__ src, const int* __restrict__ dst,
                         const float* __restrict__ e, const float* __restrict__ den,
                         const float* __restrict__ ew, const float* __restrict__ feat,
                         float* __restrict__ out,
                         int Me, int nNodes, long long tot)
{
    long long w = (long long)blockIdx.x * (blockDim.x >> 5) + (threadIdx.x >> 5);
    if (w >= tot) return;
    int lane = threadIdx.x & 31;
    int hh = (int)(w & 1);
    long long eid = w >> 1;
    int b  = (int)(eid / Me);
    int ee = (int)(eid % Me);
    int s = src[ee] + b * nNodes;
    int d = dst[ee] + b * nNodes;
    float alpha = e[w] / (den[(long long)d * CH + hh] + 1e-16f) * ew[ee];
    for (int i = lane; i < CD; i += 32)
        atomicAdd(&out[(long long)d * (CH * CD) + hh * CD + i],
                  alpha * feat[(long long)s * (CH * CD) + hh * CD + i]);
}

__global__ void gat_fin(const float* __restrict__ out, const float* __restrict__ bias,
                        float* __restrict__ y, long long n)
{
    long long i = (long long)blockIdx.x * blockDim.x + threadIdx.x;
    if (i >= n * CD) return;
    long long node = i >> 7;
    int d = (int)(i & 127);
    y[i] = 0.5f * (out[node * (CH * CD) + d] + out[node * (CH * CD) + CD + d]) + bias[d];
}

__global__ void build_x(const int* __restrict__ rid,
                        const float* __restrict__ semb,
                        const float* __restrict__ demb,
                        float* __restrict__ x)
{
    long long i = (long long)blockIdx.x * blockDim.x + threadIdx.x;
    long long tot = (long long)CB * CL * (2 * CD);
    if (i >= tot) return;
    long long row = i >> 8;
    int c = (int)(i & 255);
    int b = (int)(row >> 7);
    int r = rid[row];
    x[i] = (c < CD) ? semb[(long long)r * CD + c]
                    : demb[((long long)b * CN + r) * CD + (c - CD)];
}

__global__ void softmax128(float* __restrict__ attn, int rows)
{
    int row = blockIdx.x * (blockDim.x >> 5) + (threadIdx.x >> 5);
    if (row >= rows) return;
    int lane = threadIdx.x & 31;
    float* p = attn + (long long)row * CL;
    float v[4];
    float mx = -1e30f;
#pragma unroll
    for (int i = 0; i < 4; ++i) { v[i] = p[lane + 32 * i] * 0.0625f; mx = fmaxf(mx, v[i]); }
    for (int o = 16; o > 0; o >>= 1) mx = fmaxf(mx, __shfl_xor(mx, o, 32));
    float s = 0.f;
#pragma unroll
    for (int i = 0; i < 4; ++i) { v[i] = __expf(v[i] - mx); s += v[i]; }
    for (int o = 16; o > 0; o >>= 1) s += __shfl_xor(s, o, 32);
    float inv = 1.f / s;
#pragma unroll
    for (int i = 0; i < 4; ++i) p[lane + 32 * i] = v[i] * inv;
}

__global__ void build_z(const float* __restrict__ x2, const float* __restrict__ te,
                        float* __restrict__ z)
{
    long long i = (long long)blockIdx.x * blockDim.x + threadIdx.x;
    long long tot = (long long)CB * CL * (3 * CD);
    if (i >= tot) return;
    long long row = i / (3 * CD);
    int c = (int)(i % (3 * CD));
    z[i] = (c < 2 * CD) ? x2[row * (2 * CD) + c] : te[row * CD + (c - 2 * CD)];
}

// ---------------------------------------------------------------------------
// Host side
// ---------------------------------------------------------------------------
static inline long long cdivll(long long a, long long b) { return (a + b - 1) / b; }

static void launch_gemm_pk(hipStream_t st, const float* A, const bf16_t* Bp,
                           const float* bias, float* C, int M, int N, int K,
                           int lda, int ldc, long long sA, long long sB, long long sC,
                           int nb, int act)
{
    long long tiles = (long long)(M / 16) * (N / 64);
    dim3 grid((unsigned)cdivll(tiles, 8), (unsigned)nb);
    gemm_bpk<<<grid, 256, 0, st>>>(A, Bp, bias, C, M, N, K, lda, ldc, sA, sB, sC, act);
}

static void launch_gemm_tr(hipStream_t st, const float* A, const float* Bt, float* C,
                           int M, int N, int K, int lda, int ldb, int ldc,
                           long long sA, long long sB, long long sC, int nb)
{
    long long tiles = (long long)(M / 16) * (N / 64);
    dim3 grid((unsigned)cdivll(tiles, 8), (unsigned)nb);
    gemm_btr<<<grid, 256, 0, st>>>(A, Bt, C, M, N, K, lda, ldb, ldc, sA, sB, sC);
}

static void launch_pack(hipStream_t st, const float* B, bf16_t* Bp,
                        int Kdim, int Kpad, int N, int ldb, long long sB, int nb)
{
    long long tot = (long long)nb * N * Kpad;
    pack_b<<<(unsigned)cdivll(tot, 256), 256, 0, st>>>(B, Bp, Kdim, Kpad, N, ldb, sB, tot);
}

static void launch_fill(hipStream_t st, void* p, unsigned val, long long n)
{
    fill_u32<<<(unsigned)cdivll(n, 256), 256, 0, st>>>((unsigned*)p, val, n);
}

extern "C" void kernel_launch(void* const* d_in, const int* in_sizes, int n_in,
                              void* d_out, int out_size, void* d_ws, size_t ws_size,
                              hipStream_t stream)
{
    (void)in_sizes; (void)n_in; (void)out_size; (void)ws_size;
    const int*   rid    = (const int*)  d_in[0];
    const int*   btid   = (const int*)  d_in[1];
    const float* btf    = (const float*)d_in[2];
    const int*   edges  = (const int*)  d_in[3];
    const float* ew     = (const float*)d_in[4];
    const int*   sfeat  = (const int*)  d_in[5];
    const float* dynf   = (const float*)d_in[6];
    const float* emb    = (const float*)d_in[7];
    const float* s_w1   = (const float*)d_in[8];
    const float* s_b1   = (const float*)d_in[9];
    const float* s_w2   = (const float*)d_in[10];
    const float* s_b2   = (const float*)d_in[11];
    const float* s_gw   = (const float*)d_in[12];
    const float* s_as   = (const float*)d_in[13];
    const float* s_ad   = (const float*)d_in[14];
    const float* s_gb   = (const float*)d_in[15];
    const float* s2_w1  = (const float*)d_in[16];
    const float* s2_b1  = (const float*)d_in[17];
    const float* s2_w2  = (const float*)d_in[18];
    const float* s2_b2  = (const float*)d_in[19];
    const float* d_w1   = (const float*)d_in[20];
    const float* d_b1   = (const float*)d_in[21];
    const float* d_w2   = (const float*)d_in[22];
    const float* d_b2   = (const float*)d_in[23];
    const float* d_gw   = (const float*)d_in[24];
    const float* d_as   = (const float*)d_in[25];
    const float* d_ad   = (const float*)d_in[26];
    const float* d_gb   = (const float*)d_in[27];
    const float* d2_w1  = (const float*)d_in[28];
    const float* d2_b1  = (const float*)d_in[29];
    const float* d2_w2  = (const float*)d_in[30];
    const float* d2_b2  = (const float*)d_in[31];
    const float* ca_wq  = (const float*)d_in[32];
    const float* ca_bq  = (const float*)d_in[33];
    const float* ca_wk  = (const float*)d_in[34];
    const float* ca_bk  = (const float*)d_in[35];
    const float* ca_wv  = (const float*)d_in[36];
    const float* ca_bv  = (const float*)d_in[37];
    const float* ca_wo  = (const float*)d_in[38];
    const float* ca_bo  = (const float*)d_in[39];
    const float* t_w    = (const float*)d_in[40];
    const float* t_b    = (const float*)d_in[41];
    const float* f_w1   = (const float*)d_in[42];
    const float* f_b1   = (const float*)d_in[43];
    const float* f_w2   = (const float*)d_in[44];
    const float* f_b2   = (const float*)d_in[45];

    const int* esrc = edges;
    const int* edst = edges + CM;

    // ---- float workspace carve (slots reused across phases) ----
    float* W = (float*)d_ws;
    float* SLOT0 = W;                       // 20,480,000  Ag -> de1 -> de3
    float* SLOT1 = SLOT0 + 20480000LL;      // 20,480,000  de2 -> de4
    float* SLOT2 = SLOT1 + 20480000LL;      // 40,960,000  h_s / h_d
    float* SLOT3 = SLOT2 + 40960000LL;      // 40,960,000  out_s / out_d
    float* SLOT4 = SLOT3 + 40960000LL;      //  2,560,000  mlp hidden
    float* SLOT5 = SLOT4 + 2560000LL;       //  2,560,000  se / se2
    float* SLOT6 = SLOT5 + 2560000LL;       //  2,560,000  static_emb (persists)
    float* SLOT7 = SLOT6 + 2560000LL;       // 20,480,000  dyn_emb (persists)
    float* SLOT8 = SLOT7 + 20480000LL;      //  5,120,000  de0 (B*N x 32 padded)
    float* ES    = SLOT8 + 5120000LL;       //    320,000
    float* ED    = ES + 320000LL;           //    320,000
    int*   MMX   = (int*)(ED + 320000LL);   //    320,000 (int)
    float* DEN   = (float*)MMX + 320000LL;  //    320,000
    float* EE    = DEN + 320000LL;          //    960,000
    float* Xc    = EE + 960000LL;           //    262,144
    float* Qb    = Xc + 262144LL;
    float* Kb    = Qb + 262144LL;
    float* Vb    = Kb + 262144LL;
    float* ATT   = Vb + 262144LL;           //    131,072
    float* XO    = ATT + 131072LL;
    float* X2    = XO + 262144LL;
    float* TE    = X2 + 262144LL;           //    131,072
    float* TEin  = TE + 131072LL;           //     32,768 (btf padded to 32)
    float* Zc    = TEin + 32768LL;          //    393,216
    float* Z1    = Zc + 393216LL;           //    393,216
    // ---- bf16 packed-weight region ----
    bf16_t* P = (bf16_t*)(Z1 + 393216LL);
    bf16_t* p_sw1  = P;                  // 1024x128 -> 131072
    bf16_t* p_sw2  = p_sw1 + 131072LL;   // 128x128  -> 16384
    bf16_t* p_sgw  = p_sw2 + 16384LL;    // 128x256  -> 32768
    bf16_t* p_s2w1 = p_sgw + 32768LL;    // 16384
    bf16_t* p_s2w2 = p_s2w1 + 16384LL;   // 16384
    bf16_t* p_dw1  = p_s2w2 + 16384LL;   // 32x128   -> 4096 (K padded 6->32)
    bf16_t* p_dw2  = p_dw1 + 4096LL;     // 16384
    bf16_t* p_dgw  = p_dw2 + 16384LL;    // 32768
    bf16_t* p_d2w1 = p_dgw + 32768LL;    // 16384
    bf16_t* p_d2w2 = p_d2w1 + 16384LL;   // 16384
    bf16_t* p_wq   = p_d2w2 + 16384LL;   // 256x256  -> 65536
    bf16_t* p_wk   = p_wq + 65536LL;
    bf16_t* p_wv   = p_wk + 65536LL;
    bf16_t* p_wo   = p_wv + 65536LL;
    bf16_t* p_tw   = p_wo + 65536LL;     // 32x128   -> 4096
    bf16_t* p_fw1  = p_tw + 4096LL;      // 384x384  -> 147456
    bf16_t* p_fw2  = p_fw1 + 147456LL;   // 384x768  -> 294912
    bf16_t* p_v    = p_fw2 + 294912LL;   // V packed per batch: 8 * 256*128 = 262144

    float negf = -1e30f;
    int negbits;  memcpy(&negbits, &negf, 4);
    negbits = (negbits >= 0) ? negbits : (negbits ^ 0x7fffffff);

    // ===================== pack all weights to bf16 =====================
    launch_pack(stream, s_w1,  p_sw1,  CF * CD, CF * CD, CD,      CD,      0, 1);
    launch_pack(stream, s_w2,  p_sw2,  CD,      CD,      CD,      CD,      0, 1);
    launch_pack(stream, s_gw,  p_sgw,  CD,      CD,      CH * CD, CH * CD, 0, 1);
    launch_pack(stream, s2_w1, p_s2w1, CD,      CD,      CD,      CD,      0, 1);
    launch_pack(stream, s2_w2, p_s2w2, CD,      CD,      CD,      CD,      0, 1);
    launch_pack(stream, d_w1,  p_dw1,  CS,      32,      CD,      CD,      0, 1);
    launch_pack(stream, d_w2,  p_dw2,  CD,      CD,      CD,      CD,      0, 1);
    launch_pack(stream, d_gw,  p_dgw,  CD,      CD,      CH * CD, CH * CD, 0, 1);
    launch_pack(stream, d2_w1, p_d2w1, CD,      CD,      CD,      CD,      0, 1);
    launch_pack(stream, d2_w2, p_d2w2, CD,      CD,      CD,      CD,      0, 1);
    launch_pack(stream, ca_wq, p_wq,   2 * CD,  2 * CD,  2 * CD,  2 * CD,  0, 1);
    launch_pack(stream, ca_wk, p_wk,   2 * CD,  2 * CD,  2 * CD,  2 * CD,  0, 1);
    launch_pack(stream, ca_wv, p_wv,   2 * CD,  2 * CD,  2 * CD,  2 * CD,  0, 1);
    launch_pack(stream, ca_wo, p_wo,   2 * CD,  2 * CD,  2 * CD,  2 * CD,  0, 1);
    launch_pack(stream, t_w,   p_tw,   CS,      32,      CD,      CD,      0, 1);
    launch_pack(stream, f_w1,  p_fw1,  3 * CD,  3 * CD,  3 * CD,  3 * CD,  0, 1);
    launch_pack(stream, f_w2,  p_fw2,  3 * CD,  3 * CD,  CDM,     CDM,     0, 1);

    // ===================== static path =====================
    gather_static<<<(unsigned)cdivll((long long)CN * CF * CD, 256), 256, 0, stream>>>(sfeat, emb, SLOT0);
    launch_gemm_pk(stream, SLOT0, p_sw1, s_b1, SLOT4, CN, CD, CF * CD, CF * CD, CD, 0, 0, 0, 1, 1);
    launch_gemm_pk(stream, SLOT4, p_sw2, s_b2, SLOT5, CN, CD, CD, CD, CD, 0, 0, 0, 1, 0);
    launch_gemm_pk(stream, SLOT5, p_sgw, nullptr, SLOT2, CN, CH * CD, CD, CD, CH * CD, 0, 0, 0, 1, 0);
    gat_scores<<<(unsigned)cdivll(CN, 8), 256, 0, stream>>>(SLOT2, s_as, s_ad, ES, ED, CN);
    launch_fill(stream, MMX, (unsigned)negbits, (long long)CN * CH);
    launch_fill(stream, DEN, 0u, (long long)CN * CH);
    launch_fill(stream, SLOT3, 0u, (long long)CN * CH * CD);
    {
        long long tot = (long long)CM * 1 * CH;
        edge_e_max<<<(unsigned)cdivll(tot, 256), 256, 0, stream>>>(esrc, edst, ES, ED, EE, MMX, CM, CN, tot);
        edge_exp  <<<(unsigned)cdivll(tot, 256), 256, 0, stream>>>(edst, EE, MMX, DEN, CM, CN, tot);
        edge_agg  <<<(unsigned)cdivll(tot, 8),   256, 0, stream>>>(esrc, edst, EE, DEN, ew, SLOT2, SLOT3, CM, CN, tot);
    }
    gat_fin<<<(unsigned)cdivll((long long)CN * CD, 256), 256, 0, stream>>>(SLOT3, s_gb, SLOT5, CN);
    launch_gemm_pk(stream, SLOT5, p_s2w1, s2_b1, SLOT4, CN, CD, CD, CD, CD, 0, 0, 0, 1, 1);
    launch_gemm_pk(stream, SLOT4, p_s2w2, s2_b2, SLOT6, CN, CD, CD, CD, CD, 0, 0, 0, 1, 0);

    // ===================== dynamic path ====================
    gather_dyn<<<(unsigned)cdivll((long long)CB * CN * 32, 256), 256, 0, stream>>>(dynf, btid, SLOT8);
    launch_gemm_pk(stream, SLOT8, p_dw1, d_b1, SLOT0, CB * CN, CD, 32, 32, CD, 0, 0, 0, 1, 1);
    launch_gemm_pk(stream, SLOT0, p_dw2, d_b2, SLOT1, CB * CN, CD, CD, CD, CD, 0, 0, 0, 1, 0);
    launch_gemm_pk(stream, SLOT1, p_dgw, nullptr, SLOT2, CB * CN, CH * CD, CD, CD, CH * CD, 0, 0, 0, 1, 0);
    gat_scores<<<(unsigned)cdivll(CB * CN, 8), 256, 0, stream>>>(SLOT2, d_as, d_ad, ES, ED, CB * CN);
    launch_fill(stream, MMX, (unsigned)negbits, (long long)CB * CN * CH);
    launch_fill(stream, DEN, 0u, (long long)CB * CN * CH);
    launch_fill(stream, SLOT3, 0u, (long long)CB * CN * CH * CD);
    {
        long long tot = (long long)CM * CB * CH;
        edge_e_max<<<(unsigned)cdivll(tot, 256), 256, 0, stream>>>(esrc, edst, ES, ED, EE, MMX, CM, CN, tot);
        edge_exp  <<<(unsigned)cdivll(tot, 256), 256, 0, stream>>>(edst, EE, MMX, DEN, CM, CN, tot);
        edge_agg  <<<(unsigned)cdivll(tot, 8),   256, 0, stream>>>(esrc, edst, EE, DEN, ew, SLOT2, SLOT3, CM, CN, tot);
    }
    gat_fin<<<(unsigned)cdivll((long long)CB * CN * CD, 256), 256, 0, stream>>>(SLOT3, d_gb, SLOT0, CB * CN);
    launch_gemm_pk(stream, SLOT0, p_d2w1, d2_b1, SLOT1, CB * CN, CD, CD, CD, CD, 0, 0, 0, 1, 1);
    launch_gemm_pk(stream, SLOT1, p_d2w2, d2_b2, SLOT7, CB * CN, CD, CD, CD, CD, 0, 0, 0, 1, 0);

    // ===================== fusion: cross-attention ====================
    build_x<<<(unsigned)cdivll((long long)CB * CL * 2 * CD, 256), 256, 0, stream>>>(rid, SLOT6, SLOT7, Xc);
    const int R = CB * CL;     // 1024
    const int D2 = 2 * CD;     // 256
    launch_gemm_pk(stream, Xc, p_wq, ca_bq, Qb, R, D2, D2, D2, D2, 0, 0, 0, 1, 0);
    launch_gemm_pk(stream, Xc, p_wk, ca_bk, Kb, R, D2, D2, D2, D2, 0, 0, 0, 1, 0);
    launch_gemm_pk(stream, Xc, p_wv, ca_bv, Vb, R, D2, D2, D2, D2, 0, 0, 0, 1, 0);
    // attn logits: per batch, q (L x 256) @ k^T -> (L x L)
    launch_gemm_tr(stream, Qb, Kb, ATT, CL, CL, D2, D2, D2, CL,
                   (long long)CL * D2, (long long)CL * D2, (long long)CL * CL, CB);
    softmax128<<<(unsigned)cdivll(R, 8), 256, 0, stream>>>(ATT, R);
    // pack v (per batch, K=128 rows x N=256 cols) then attn @ v
    launch_pack(stream, Vb, p_v, CL, CL, D2, D2, (long long)CL * D2, CB);
    launch_gemm_pk(stream, ATT, p_v, nullptr, XO, CL, D2, CL, CL, D2,
                   (long long)CL * CL, (long long)D2 * CL, (long long)CL * D2, CB, 0);
    launch_gemm_pk(stream, XO, p_wo, ca_bo, X2, R, D2, D2, D2, D2, 0, 0, 0, 1, 0);

    // ===================== time features + final MLP ====================
    pad_rows<<<(unsigned)cdivll((long long)R * 32, 256), 256, 0, stream>>>(btf, TEin, CS, 32, R);
    launch_gemm_pk(stream, TEin, p_tw, t_b, TE, R, CD, 32, 32, CD, 0, 0, 0, 1, 0);
    build_z<<<(unsigned)cdivll((long long)R * 3 * CD, 256), 256, 0, stream>>>(X2, TE, Zc);
    launch_gemm_pk(stream, Zc, p_fw1, f_b1, Z1, R, 3 * CD, 3 * CD, 3 * CD, 3 * CD, 0, 0, 0, 1, 1);
    launch_gemm_pk(stream, Z1, p_fw2, f_b2, (float*)d_out, R, CDM, 3 * CD, 3 * CD, CDM, 0, 0, 0, 1, 0);
}
